// Scaled_Dot_Product_Attention_35931696398997
// MI455X (gfx1250) — compile-verified, matrix operand order checked by reading, not run on hardware
//
#include <hip/hip_runtime.h>
#include <hip/hip_bf16.h>

typedef __attribute__((ext_vector_type(16))) _Float16 v16h;
typedef __attribute__((ext_vector_type(8)))  _Float16 v8h;
typedef __attribute__((ext_vector_type(8)))  float    v8f;

#define B_ 8
#define L_ 2048
#define D_ 512
#define QT 16          // queries per workgroup (one WMMA M tile)
#define KT 128         // keys per iteration (8 waves x 16)
#define NITER (L_/KT)

// Build a 16x32-K f16 WMMA fragment from a row-major source row.
// Lane layout (both A and B operands): index = lane&15, halves:
//   K = 16*(h>>3) + 8*(lane>>4) + (h&7)  -> two contiguous 8-half (16B) runs.
__device__ __forceinline__ v16h load_frag(const _Float16* base, int hi) {
  v8h lo = *(const v8h*)(base + 8 * hi);
  v8h up = *(const v8h*)(base + 16 + 8 * hi);
  v16h r;
#pragma unroll
  for (int i = 0; i < 8; ++i) { r[i] = lo[i]; r[i + 8] = up[i]; }
  return r;
}

// Pre-pass: f32->f16 conversion (Q pre-scaled), V transposed to [b][d][l],
// per-key relation biases bias = scale * (K . rel).
__global__ __launch_bounds__(128) void prep_kernel(
    const float* __restrict__ Q, const float* __restrict__ K,
    const float* __restrict__ V, const float* __restrict__ adj_rel,
    const float* __restrict__ com_rel, const float* __restrict__ scale_p,
    _Float16* __restrict__ Qh, _Float16* __restrict__ Kh,
    _Float16* __restrict__ Vt, float* __restrict__ bias_a,
    float* __restrict__ bias_c) {
  const int row = blockIdx.x;            // b*L + l
  const int b = row / L_;
  const int l = row % L_;
  const int tid = threadIdx.x;
  const float scale = scale_p[0];
  const size_t base = (size_t)row * D_;
  float ka = 0.f, kc = 0.f;
  for (int d = tid; d < D_; d += 128) {
    float q = Q[base + d];
    Qh[base + d] = (_Float16)(q * scale);
    float k = K[base + d];
    Kh[base + d] = (_Float16)k;
    ka += k * adj_rel[d];
    kc += k * com_rel[d];
    float v = V[base + d];
    Vt[((size_t)b * D_ + d) * L_ + l] = (_Float16)v;
  }
  __shared__ float ra[128], rc[128];
  ra[tid] = ka; rc[tid] = kc;
  __syncthreads();
  for (int s = 64; s > 0; s >>= 1) {
    if (tid < s) { ra[tid] += ra[tid + s]; rc[tid] += rc[tid + s]; }
    __syncthreads();
  }
  if (tid == 0) { bias_a[row] = scale * ra[0]; bias_c[row] = scale * rc[0]; }
}

__global__ __launch_bounds__(256) void attn_kernel(
    const _Float16* __restrict__ Qh, const _Float16* __restrict__ Kh,
    const _Float16* __restrict__ Vt, const float* __restrict__ bias_a,
    const float* __restrict__ bias_c, const float* __restrict__ adj_mask,
    const float* __restrict__ com_mask, float* __restrict__ out) {
  const int tid = threadIdx.x;
  const int lane = tid & 31;
  const int w = tid >> 5;               // wave 0..7
  const int l15 = lane & 15;
  const int hi = lane >> 4;
  const int blk = blockIdx.x;
  const int b = blk >> 7;               // / (L/QT) = /128
  const int qbase = (blk & 127) * QT;
  const int dw = w * 64;                // this wave's output D slice

  __shared__ __align__(16) _Float16 Qs[QT][D_];          // 16 KB
  __shared__ float Tl[2][QT][KT];                        // 16 KB logits
  __shared__ __align__(16) _Float16 Pl[2][QT][KT];       // 8 KB probs
  __shared__ float stat_m[32], stat_l[32], stat_scale[32], stat_mcur[32];

  // stage Q tile (already scaled)
  for (int i = tid; i < QT * D_; i += 256) {
    int r = i >> 9, d = i & (D_ - 1);
    Qs[r][d] = Qh[((size_t)b * L_ + qbase + r) * D_ + d];
  }
  if (tid < 32) { stat_m[tid] = -3.0e38f; stat_l[tid] = 0.f; }
  __syncthreads();

  v8f acc[2][4];
#pragma unroll
  for (int br = 0; br < 2; ++br)
#pragma unroll
    for (int nt = 0; nt < 4; ++nt)
#pragma unroll
      for (int v = 0; v < 8; ++v) acc[br][nt][v] = 0.f;

  for (int it = 0; it < NITER; ++it) {
    const int kbase = it * KT;
    const int krow = kbase + w * 16 + l15;

    // ---- S tile: Q(16x512) x K^T for this wave's 16 keys ----
    const _Float16* kp = Kh + ((size_t)b * L_ + krow) * D_;
    v8f s;
#pragma unroll
    for (int v = 0; v < 8; ++v) s[v] = 0.f;
#pragma unroll
    for (int dc = 0; dc < 16; ++dc) {
      v16h a = load_frag(&Qs[l15][dc * 32], hi);
      v16h kb = load_frag(kp + dc * 32, hi);
      s = __builtin_amdgcn_wmma_f32_16x16x32_f16(false, a, false, kb,
                                                 (short)0, s, false, false);
    }
    // add per-key branch biases, spill logits to LDS (C layout: row=v+8*hi)
    const int kcol = kbase + w * 16 + l15;
    float ba = bias_a[b * L_ + kcol];
    float bc = bias_c[b * L_ + kcol];
    const int cl = w * 16 + l15;
#pragma unroll
    for (int v = 0; v < 8; ++v) {
      int r = v + 8 * hi;
      Tl[0][r][cl] = s[v] + ba;
      Tl[1][r][cl] = s[v] + bc;
    }
    __syncthreads();

    // ---- online-softmax row stats: 32 (row,branch) pairs, 8 threads each
    {
      int pair = tid >> 3;
      int sub = tid & 7;
      int br = pair >> 4;
      int row = pair & 15;
      const float* trow = &Tl[br][row][0];
      float mx = -3.0e38f;
#pragma unroll
      for (int c = 0; c < 16; ++c) mx = fmaxf(mx, trow[sub * 16 + c]);
#pragma unroll
      for (int off = 1; off < 8; off <<= 1)
        mx = fmaxf(mx, __shfl_xor(mx, off, 32));
      float m_old = stat_m[pair];
      float m_new = fmaxf(m_old, mx);
      float es = 0.f;
#pragma unroll
      for (int c = 0; c < 16; ++c) es += __expf(trow[sub * 16 + c] - m_new);
#pragma unroll
      for (int off = 1; off < 8; off <<= 1) es += __shfl_xor(es, off, 32);
      if (sub == 0) {
        float sc = __expf(m_old - m_new);
        stat_scale[pair] = sc;
        stat_mcur[pair] = m_new;
        stat_m[pair] = m_new;
        stat_l[pair] = stat_l[pair] * sc + es;
      }
    }
    __syncthreads();

    // ---- rescale running accumulators ----
#pragma unroll
    for (int br = 0; br < 2; ++br)
#pragma unroll
      for (int v = 0; v < 8; ++v) {
        float sc = stat_scale[br * 16 + v + 8 * hi];
#pragma unroll
        for (int nt = 0; nt < 4; ++nt) acc[br][nt][v] *= sc;
      }

    // ---- P = exp(t - m) * mask, converted to f16 (masks streamed once) ----
#pragma unroll
    for (int rep = 0; rep < 16; ++rep) {
      int idx = rep * 256 + tid;        // 2*16*128 = 4096 elements
      int br = idx >> 11;
      int rem = idx & 2047;
      int row = rem >> 7;
      int col = rem & 127;
      const float* mp = br ? com_mask : adj_mask;
      float mv = mp[((size_t)b * L_ + qbase + row) * L_ + kbase + col];
      float p = __expf(Tl[br][row][col] - stat_mcur[br * 16 + row]) * mv;
      Pl[br][row][col] = (_Float16)p;
    }
    __syncthreads();

    // ---- acc += P x V for this wave's 64-wide D slice ----
#pragma unroll
    for (int br = 0; br < 2; ++br) {
#pragma unroll
      for (int kc = 0; kc < 4; ++kc) {
        v16h a = load_frag(&Pl[br][l15][kc * 32], hi);
#pragma unroll
        for (int nt = 0; nt < 4; ++nt) {
          const _Float16* vp =
              Vt + ((size_t)b * D_ + dw + nt * 16 + l15) * L_ + kbase + kc * 32;
          v16h bf = load_frag(vp, hi);
          acc[br][nt] = __builtin_amdgcn_wmma_f32_16x16x32_f16(
              false, a, false, bf, (short)0, acc[br][nt], false, false);
        }
      }
    }
    __syncthreads();
  }

  // ---- epilogue: out = acc_a/l_a + acc_c/l_c ----
  float* op = out + ((size_t)b * L_ + qbase) * D_;
#pragma unroll
  for (int v = 0; v < 8; ++v) {
    int r = v + 8 * hi;
    float inva = 1.0f / stat_l[r];
    float invc = 1.0f / stat_l[16 + r];
#pragma unroll
    for (int nt = 0; nt < 4; ++nt) {
      int d = dw + nt * 16 + l15;
      op[(size_t)r * D_ + d] = acc[0][nt][v] * inva + acc[1][nt][v] * invc;
    }
  }
}

extern "C" void kernel_launch(void* const* d_in, const int* in_sizes, int n_in,
                              void* d_out, int out_size, void* d_ws,
                              size_t ws_size, hipStream_t stream) {
  (void)in_sizes; (void)n_in; (void)out_size; (void)ws_size;
  const float* Q        = (const float*)d_in[0];
  const float* K        = (const float*)d_in[1];
  const float* V        = (const float*)d_in[2];
  const float* adj_mask = (const float*)d_in[3];
  const float* com_mask = (const float*)d_in[4];
  const float* adj_rel  = (const float*)d_in[5];
  const float* com_rel  = (const float*)d_in[6];
  const float* scale_p  = (const float*)d_in[7];

  const size_t BLD = (size_t)B_ * L_ * D_;
  _Float16* Qh = (_Float16*)d_ws;
  _Float16* Kh = Qh + BLD;
  _Float16* Vt = Kh + BLD;
  float* bias_a = (float*)(Vt + BLD);
  float* bias_c = bias_a + (size_t)B_ * L_;

  prep_kernel<<<B_ * L_, 128, 0, stream>>>(Q, K, V, adj_rel, com_rel, scale_p,
                                           Qh, Kh, Vt, bias_a, bias_c);
  attn_kernel<<<B_ * (L_ / QT), 256, 0, stream>>>(
      Qh, Kh, Vt, bias_a, bias_c, adj_mask, com_mask, (float*)d_out);
}
